// InformerSpatialClassify_9131100471627
// MI455X (gfx1250) — compile-verified, matrix operand order checked by reading, not run on hardware
//
#include <hip/hip_runtime.h>
#include <hip/hip_bf16.h>
#include <math.h>

// ---------------------------------------------------------------------------
// CDNA5 (gfx1250) Informer forward.  wave32, WMMA f16->f32, LDS-tiled GEMM.
// ---------------------------------------------------------------------------

typedef __attribute__((ext_vector_type(16))) _Float16 v16h;
typedef __attribute__((ext_vector_type(8)))  float    v8f;

// Per-lane K-base for the 16-bit A/B fragment layout (ISA 7.12.2):
// VGPR v<4 : K = 2v (+8 for lanes 16..31); v>=4 : K = 16+2(v-4) (+8 hi half)
__device__ __forceinline__ int frag_kbase(int v, int hi) {
  int base = (v < 4) ? (2 * v) : (16 + 2 * (v - 4));
  return base + (hi ? 8 : 0);
}

__device__ __forceinline__ float gelu_exact(float x) {
  return 0.5f * x * (1.0f + erff(x * 0.70710678118654752f));
}

__device__ __forceinline__ v8f wmma16(v16h a, v16h b, v8f c) {
  return __builtin_amdgcn_wmma_f32_16x16x32_f16(false, a, false, b, (short)0, c,
                                                false, false);
}

// ---------------------------------------------------------------------------
// LDS-tiled GEMM: C[M,N] = A[M,K] * B + epilogue.
// Block = 256 threads (8 waves) computes a 64x64 tile; each wave computes two
// 16x16 WMMA subtiles sharing one A fragment.  A/B staged as f16 in LDS.
// A: f32 row-major (optionally circular row-remapped for conv taps).
// B: f32 with arbitrary strides: B[k,n] = Bm[boff + k*bks + n*bns].
// flags: 1=+bias[n], 2=GELU, 4=+resid[m,n], 8=accumulate into C.
// Requires M%64==0, N%64==0, K%32==0 (true for every GEMM in this model).
// ---------------------------------------------------------------------------
#define GBM 64
#define GBN 64
#define GBK 32

__global__ __launch_bounds__(256)
void gemm_wmma_f16(const float* __restrict__ A,
                   const float* __restrict__ Bm,
                   float* __restrict__ C,
                   const float* __restrict__ bias,
                   const float* __restrict__ resid,
                   int M, int N, int K,
                   int bks, int bns, int boff,
                   int convL, int convShift, int flags) {
  __shared__ _Float16 sA[GBM * GBK];  // [m][k]
  __shared__ _Float16 sB[GBK * GBN];  // [k][n]

  int tid = threadIdx.x;
  int lane = tid & 31, wv = tid >> 5;
  int hi = lane >> 4, l15 = lane & 15;
  int n0 = blockIdx.x * GBN;
  int m0 = blockIdx.y * GBM;
  int sm = (wv & 3) * 16;   // wave's subtile row offset in block
  int sn = (wv >> 2) * 32;  // wave's 2 subtiles at sn, sn+16

  v8f acc0 = {}, acc1 = {};

  for (int k0 = 0; k0 < K; k0 += GBK) {
    // stage A (64x32) -> f16 LDS, with circular conv row remap
    for (int i = tid; i < GBM * GBK; i += 256) {
      int mm = i >> 5;    // /GBK
      int kk = i & 31;
      int r = m0 + mm;
      int asrc = r;
      if (convL > 0) {
        int bb = r / convL;
        int l = r - bb * convL;
        int ls = l + convShift;
        if (ls < 0) ls += convL;
        if (ls >= convL) ls -= convL;
        asrc = bb * convL + ls;
      }
      const float* ap = A + (size_t)asrc * K + k0 + kk;
      sA[i] = (_Float16)(*ap);
      if (k0 + GBK < K) __builtin_prefetch(ap + GBK, 0, 1);  // global_prefetch
    }
    // stage B (32x64) -> f16 LDS
    for (int i = tid; i < GBK * GBN; i += 256) {
      int kk = i >> 6;    // /GBN
      int nn = i & 63;
      const float* bp = Bm + (size_t)boff + (size_t)(k0 + kk) * bks +
                        (size_t)(n0 + nn) * bns;
      sB[i] = (_Float16)(*bp);
      if (k0 + GBK < K) __builtin_prefetch(bp + (size_t)GBK * bks, 0, 1);
    }
    __syncthreads();

    v16h af, bf0, bf1;
    const _Float16* Arow = sA + (sm + l15) * GBK;
#pragma unroll
    for (int v = 0; v < 8; ++v) {
      int kk = frag_kbase(v, hi);
      af[2 * v]     = Arow[kk];
      af[2 * v + 1] = Arow[kk + 1];
      bf0[2 * v]     = sB[kk * GBN + sn + l15];
      bf0[2 * v + 1] = sB[(kk + 1) * GBN + sn + l15];
      bf1[2 * v]     = sB[kk * GBN + sn + 16 + l15];
      bf1[2 * v + 1] = sB[(kk + 1) * GBN + sn + 16 + l15];
    }
    acc0 = wmma16(af, bf0, acc0);
    acc1 = wmma16(af, bf1, acc1);
    __syncthreads();
  }

  int col0 = n0 + sn + l15;
  int col1 = col0 + 16;
  float bv0 = (flags & 1) ? bias[col0] : 0.0f;
  float bv1 = (flags & 1) ? bias[col1] : 0.0f;
#pragma unroll
  for (int rr = 0; rr < 8; ++rr) {
    int row = m0 + sm + rr + (hi ? 8 : 0);
    float v0 = acc0[rr] + bv0;
    float v1 = acc1[rr] + bv1;
    if (flags & 2) { v0 = gelu_exact(v0); v1 = gelu_exact(v1); }
    size_t i0 = (size_t)row * N + col0;
    size_t i1 = (size_t)row * N + col1;
    if (flags & 8) { v0 += C[i0]; v1 += C[i1]; }
    if (flags & 4) { v0 += resid[i0]; v1 += resid[i1]; }
    C[i0] = v0;
    C[i1] = v1;
  }
}

// ---------------------------------------------------------------------------
// Fused temporal attention, one workgroup per (b,h).  dh = 64.
// mode 0: full softmax attention (cross)
// mode 1: ProbSparse non-masked (context = mean(V), top-u rows get softmax)
// mode 2: ProbSparse causal    (context = cumsum(V), top-u rows causal softmax)
// Q/K/V layout: (B, L, H, 64).  Output BLHD (out_bhld=0) or BHLD (=1).
// ---------------------------------------------------------------------------
__global__ __launch_bounds__(256)
void attn_temporal_k(const float* __restrict__ Qg,
                     const float* __restrict__ Kg,
                     const float* __restrict__ Vg,
                     float* __restrict__ Og,
                     int H, int Lq, int Lk, int LqP, int LkP,
                     int mode, int u, float scale, int out_bhld) {
  extern __shared__ float sm[];
  float* sQ  = sm;               // LqP*64
  float* sK  = sQ + LqP * 64;    // LkP*64
  float* sV  = sK + LkP * 64;    // LkP*64
  float* sS  = sV + LkP * 64;    // LqP*LkP
  float* sCB = sS + LqP * LkP;   // LqP*64 (cumsum base, mode 2)
  float* sM  = sCB + LqP * 64;   // LqP
  float* sVm = sM + LqP;         // 64
  int*   sSel = (int*)(sVm + 64);  // LqP

  int bh = blockIdx.x;
  int b = bh / H, h = bh % H;
  int tid = threadIdx.x, nthr = blockDim.x;
  int lane = tid & 31, wv = tid >> 5, nw = nthr >> 5;
  int hi = lane >> 4, l15 = lane & 15;

  for (int i = tid; i < LqP * 64; i += nthr) {
    int l = i >> 6, d = i & 63;
    sQ[i] = (l < Lq) ? Qg[(((size_t)b * Lq + l) * H + h) * 64 + d] : 0.f;
  }
  for (int i = tid; i < LkP * 64; i += nthr) {
    int l = i >> 6, d = i & 63;
    float kv = 0.f, vv = 0.f;
    if (l < Lk) {
      kv = Kg[(((size_t)b * Lk + l) * H + h) * 64 + d];
      vv = Vg[(((size_t)b * Lk + l) * H + h) * 64 + d];
    }
    sK[i] = kv;
    sV[i] = vv;
  }
  for (int i = tid; i < LqP; i += nthr) sSel[i] = 0;
  __syncthreads();

  // S = Q K^T (raw, unscaled)
  int tilesM = LqP >> 4, tilesN = LkP >> 4;
  for (int t = wv; t < tilesM * tilesN; t += nw) {
    int ti = t / tilesN, tj = t - ti * tilesN;
    int m0 = ti << 4, n0 = tj << 4;
    const float* Ar = sQ + (m0 + l15) * 64;
    const float* Br = sK + (n0 + l15) * 64;
    v8f acc = {};
#pragma unroll
    for (int k0 = 0; k0 < 64; k0 += 32) {
      v16h af, bf;
#pragma unroll
      for (int v = 0; v < 8; ++v) {
        int kk = k0 + frag_kbase(v, hi);
        af[2 * v] = (_Float16)Ar[kk]; af[2 * v + 1] = (_Float16)Ar[kk + 1];
        bf[2 * v] = (_Float16)Br[kk]; bf[2 * v + 1] = (_Float16)Br[kk + 1];
      }
      acc = wmma16(af, bf, acc);
    }
#pragma unroll
    for (int rr = 0; rr < 8; ++rr) {
      int row = m0 + rr + (hi ? 8 : 0);
      sS[row * LkP + n0 + l15] = acc[rr];
    }
  }
  __syncthreads();

  if (mode == 1) {  // mean of V
    for (int d = tid; d < 64; d += nthr) {
      float s = 0.f;
      for (int j = 0; j < Lk; ++j) s += sV[j * 64 + d];
      sVm[d] = s / (float)Lk;
    }
  } else if (mode == 2) {  // cumulative sum of V
    for (int i = tid; i < Lq * 64; i += nthr) {
      int l = i >> 6, d = i & 63;
      float s = 0.f;
      for (int j = 0; j <= l; ++j) s += sV[j * 64 + d];
      sCB[l * 64 + d] = s;
    }
  }
  if (mode != 0) {  // sparsity measure M = max - mean over keys
    for (int l = tid; l < Lq; l += nthr) {
      const float* Sr = sS + l * LkP;
      float mx = -1e30f, sum = 0.f;
      for (int j = 0; j < Lk; ++j) { float v = Sr[j]; mx = fmaxf(mx, v); sum += v; }
      sM[l] = mx - sum / (float)Lk;
    }
  }
  __syncthreads();
  if (mode != 0 && tid == 0) {  // top-u selection
    for (int it = 0; it < u; ++it) {
      int best = 0; float bvv = -1e30f;
      for (int l = 0; l < Lq; ++l) if (sM[l] > bvv) { bvv = sM[l]; best = l; }
      sSel[best] = 1;
      sM[best] = -1e30f;
    }
  }
  __syncthreads();

  // softmax rows (scale, optional causal mask); zero padded cols/rows
  for (int l = tid; l < Lq; l += nthr) {
    float* Sr = sS + l * LkP;
    float mx = -1e30f;
    for (int j = 0; j < Lk; ++j) {
      float v = Sr[j] * scale;
      if (mode == 2 && j > l) v = -1e9f;
      Sr[j] = v;
      mx = fmaxf(mx, v);
    }
    float sum = 0.f;
    for (int j = 0; j < Lk; ++j) { float e = expf(Sr[j] - mx); Sr[j] = e; sum += e; }
    float inv = 1.f / sum;
    for (int j = 0; j < Lk; ++j) Sr[j] *= inv;
    for (int j = Lk; j < LkP; ++j) Sr[j] = 0.f;
  }
  for (int l = Lq + tid; l < LqP; l += nthr) {
    float* Sr = sS + l * LkP;
    for (int j = 0; j < LkP; ++j) Sr[j] = 0.f;
  }
  __syncthreads();

  // O = P V, with fallback context for non-selected rows
  for (int t = wv; t < tilesM * 4; t += nw) {
    int ti = t >> 2, tj = t & 3;
    int m0 = ti << 4, n0 = tj << 4;
    const float* Ar = sS + (m0 + l15) * LkP;
    v8f acc = {};
    for (int k0 = 0; k0 < LkP; k0 += 32) {
      v16h af, bf;
#pragma unroll
      for (int v = 0; v < 8; ++v) {
        int kk = k0 + frag_kbase(v, hi);
        af[2 * v] = (_Float16)Ar[kk]; af[2 * v + 1] = (_Float16)Ar[kk + 1];
        bf[2 * v]     = (_Float16)sV[kk * 64 + n0 + l15];
        bf[2 * v + 1] = (_Float16)sV[(kk + 1) * 64 + n0 + l15];
      }
      acc = wmma16(af, bf, acc);
    }
#pragma unroll
    for (int rr = 0; rr < 8; ++rr) {
      int row = m0 + rr + (hi ? 8 : 0);
      if (row >= Lq) continue;
      int d = n0 + l15;
      float o = acc[rr];
      if (mode == 1 && !sSel[row]) o = sVm[d];
      if (mode == 2 && !sSel[row]) o = sCB[row * 64 + d];
      size_t oidx = out_bhld
          ? ((((size_t)b * H + h) * Lq + row) * 64 + d)
          : ((((size_t)b * Lq + row) * H + h) * 64 + d);
      Og[oidx] = o;
    }
  }
}

// ---------------------------------------------------------------------------
// Spatial ProbSparse attention: L=512, dh=16 (padded to 32), per (b,h).
// ---------------------------------------------------------------------------
__global__ __launch_bounds__(256)
void attn_spatial_k(const float* __restrict__ Qg,
                    const float* __restrict__ Kg,
                    const float* __restrict__ Vg,
                    float* __restrict__ Og,
                    int H, int L, int u, float scale) {
  extern __shared__ float sm[];
  float* sQ  = sm;               // L*32
  float* sK  = sQ + L * 32;      // L*32
  float* sV  = sK + L * 32;      // L*16
  float* sS  = sV + L * 16;      // 32*L
  float* sQs = sS + 32 * L;      // 32*32
  float* sM  = sQs + 1024;       // L
  float* sVm = sM + L;           // 16
  int* sList = (int*)(sVm + 16); // L
  int* sFlag = sList + L;        // L

  int bh = blockIdx.x;
  int b = bh / H, h = bh % H;
  int tid = threadIdx.x, nthr = blockDim.x;
  int lane = tid & 31, wv = tid >> 5, nw = nthr >> 5;
  int hi = lane >> 4, l15 = lane & 15;
  int tilesN = L >> 4;

  for (int i = tid; i < L * 32; i += nthr) {
    int l = i >> 5, d = i & 31;
    float q = 0.f, k = 0.f;
    if (d < 16) {
      q = Qg[(((size_t)b * L + l) * H + h) * 16 + d];
      k = Kg[(((size_t)b * L + l) * H + h) * 16 + d];
    }
    sQ[i] = q;
    sK[i] = k;
  }
  for (int i = tid; i < L * 16; i += nthr) {
    int l = i >> 4, d = i & 15;
    sV[i] = Vg[(((size_t)b * L + l) * H + h) * 16 + d];
  }
  for (int i = tid; i < L; i += nthr) sFlag[i] = 0;
  __syncthreads();

  for (int d = tid; d < 16; d += nthr) {
    float s = 0.f;
    for (int j = 0; j < L; ++j) s += sV[j * 16 + d];
    sVm[d] = s / (float)L;
  }
  __syncthreads();

  // phase 1: sparsity measure per 32-row score blocks
  for (int blk = 0; blk < L / 32; ++blk) {
    for (int t = wv; t < 2 * tilesN; t += nw) {
      int ti = t / tilesN, tj = t - ti * tilesN;
      int m0 = ti << 4, n0 = tj << 4;
      const float* Ar = sQ + (blk * 32 + m0 + l15) * 32;
      const float* Br = sK + (n0 + l15) * 32;
      v8f acc = {};
      v16h af, bf;
#pragma unroll
      for (int v = 0; v < 8; ++v) {
        int kk = frag_kbase(v, hi);
        af[2 * v] = (_Float16)Ar[kk]; af[2 * v + 1] = (_Float16)Ar[kk + 1];
        bf[2 * v] = (_Float16)Br[kk]; bf[2 * v + 1] = (_Float16)Br[kk + 1];
      }
      acc = wmma16(af, bf, acc);
#pragma unroll
      for (int rr = 0; rr < 8; ++rr) {
        int row = m0 + rr + (hi ? 8 : 0);
        sS[row * L + n0 + l15] = acc[rr];
      }
    }
    __syncthreads();
    for (int l = tid; l < 32; l += nthr) {
      const float* Sr = sS + l * L;
      float mx = -1e30f, sum = 0.f;
      for (int j = 0; j < L; ++j) { float v = Sr[j]; mx = fmaxf(mx, v); sum += v; }
      sM[blk * 32 + l] = mx - sum / (float)L;
    }
    __syncthreads();
  }

  if (tid == 0) {
    for (int it = 0; it < u; ++it) {
      int best = 0; float bvv = -1e30f;
      for (int l = 0; l < L; ++l) if (sM[l] > bvv) { bvv = sM[l]; best = l; }
      sList[it] = best;
      sFlag[best] = 1;
      sM[best] = -1e30f;
    }
  }
  __syncthreads();

  // base context = mean(V) for non-selected rows
  for (int i = tid; i < L * 16; i += nthr) {
    int l = i >> 4, d = i & 15;
    if (!sFlag[l]) Og[(((size_t)b * L + l) * H + h) * 16 + d] = sVm[d];
  }
  __syncthreads();

  // phase 2: softmax attention for selected queries, 32 at a time
  int ngrp = (u + 31) / 32;
  for (int g = 0; g < ngrp; ++g) {
    int cnt = u - g * 32; if (cnt > 32) cnt = 32;
    for (int i = tid; i < 1024; i += nthr) {
      int r = i >> 5, d = i & 31;
      sQs[i] = (r < cnt) ? sQ[sList[g * 32 + r] * 32 + d] : 0.f;
    }
    __syncthreads();
    for (int t = wv; t < 2 * tilesN; t += nw) {
      int ti = t / tilesN, tj = t - ti * tilesN;
      int m0 = ti << 4, n0 = tj << 4;
      const float* Ar = sQs + (m0 + l15) * 32;
      const float* Br = sK + (n0 + l15) * 32;
      v8f acc = {};
      v16h af, bf;
#pragma unroll
      for (int v = 0; v < 8; ++v) {
        int kk = frag_kbase(v, hi);
        af[2 * v] = (_Float16)Ar[kk]; af[2 * v + 1] = (_Float16)Ar[kk + 1];
        bf[2 * v] = (_Float16)Br[kk]; bf[2 * v + 1] = (_Float16)Br[kk + 1];
      }
      acc = wmma16(af, bf, acc);
#pragma unroll
      for (int rr = 0; rr < 8; ++rr) {
        int row = m0 + rr + (hi ? 8 : 0);
        sS[row * L + n0 + l15] = acc[rr];
      }
    }
    __syncthreads();
    for (int r = tid; r < cnt; r += nthr) {
      float* Sr = sS + r * L;
      float mx = -1e30f;
      for (int j = 0; j < L; ++j) { float v = Sr[j] * scale; Sr[j] = v; mx = fmaxf(mx, v); }
      float sum = 0.f;
      for (int j = 0; j < L; ++j) { float e = expf(Sr[j] - mx); Sr[j] = e; sum += e; }
      float inv = 1.f / sum;
      for (int j = 0; j < L; ++j) Sr[j] *= inv;
    }
    for (int r = cnt + tid; r < 32; r += nthr) {
      float* Sr = sS + r * L;
      for (int j = 0; j < L; ++j) Sr[j] = 0.f;
    }
    __syncthreads();
    for (int t = wv; t < 2; t += nw) {
      int m0 = t << 4;
      const float* Ar = sS + (m0 + l15) * L;
      v8f acc = {};
      for (int k0 = 0; k0 < L; k0 += 32) {
        v16h af, bf;
#pragma unroll
        for (int v = 0; v < 8; ++v) {
          int kk = k0 + frag_kbase(v, hi);
          af[2 * v] = (_Float16)Ar[kk]; af[2 * v + 1] = (_Float16)Ar[kk + 1];
          bf[2 * v]     = (_Float16)sV[kk * 16 + l15];
          bf[2 * v + 1] = (_Float16)sV[(kk + 1) * 16 + l15];
        }
        acc = wmma16(af, bf, acc);
      }
#pragma unroll
      for (int rr = 0; rr < 8; ++rr) {
        int row = m0 + rr + (hi ? 8 : 0);
        if (row < cnt) {
          int gl = sList[g * 32 + row];
          Og[(((size_t)b * L + gl) * H + h) * 16 + l15] = acc[rr];
        }
      }
    }
    __syncthreads();
  }
}

// ---------------------------------------------------------------------------
// Small helper kernels
// ---------------------------------------------------------------------------
__global__ void add_pe_k(float* __restrict__ x, int total, int L, int D) {
  int i = blockIdx.x * 256 + threadIdx.x;
  if (i >= total) return;
  int c = i % D;
  int l = (i / D) % L;
  int pi = c >> 1;
  float e = expf(-(float)(2 * pi) * (logf(10000.0f) / (float)D));
  float a = (float)l * e;
  x[i] += (c & 1) ? cosf(a) : sinf(a);
}

__global__ void layer_norm_k(const float* __restrict__ x, float* __restrict__ y,
                             const float* __restrict__ g, const float* __restrict__ b,
                             int D) {
  __shared__ float red[128];
  int row = blockIdx.x, tid = threadIdx.x;
  const float* xr = x + (size_t)row * D;
  float s = 0.f;
  for (int i = tid; i < D; i += 128) s += xr[i];
  red[tid] = s; __syncthreads();
  for (int o = 64; o; o >>= 1) { if (tid < o) red[tid] += red[tid + o]; __syncthreads(); }
  float mean = red[0] / (float)D; __syncthreads();
  float s2 = 0.f;
  for (int i = tid; i < D; i += 128) { float d = xr[i] - mean; s2 += d * d; }
  red[tid] = s2; __syncthreads();
  for (int o = 64; o; o >>= 1) { if (tid < o) red[tid] += red[tid + o]; __syncthreads(); }
  float inv = rsqrtf(red[0] / (float)D + 1e-5f);
  float* yr = y + (size_t)row * D;
  for (int i = tid; i < D; i += 128) yr[i] = (xr[i] - mean) * inv * g[i] + b[i];
}

__global__ void bn_stats_k(const float* __restrict__ x, float* __restrict__ stats,
                           int rows, int C) {
  __shared__ float s1[256], s2[256];
  int c = blockIdx.x, tid = threadIdx.x;
  float a = 0.f, b = 0.f;
  for (int r = tid; r < rows; r += 256) {
    float v = x[(size_t)r * C + c];
    a += v; b += v * v;
  }
  s1[tid] = a; s2[tid] = b; __syncthreads();
  for (int o = 128; o; o >>= 1) {
    if (tid < o) { s1[tid] += s1[tid + o]; s2[tid] += s2[tid + o]; }
    __syncthreads();
  }
  if (tid == 0) {
    float m = s1[0] / (float)rows;
    stats[c] = m;
    stats[C + c] = s2[0] / (float)rows - m * m;
  }
}

__global__ void bn_elu_k(float* __restrict__ x, const float* __restrict__ stats,
                         const float* __restrict__ g, const float* __restrict__ b,
                         int total, int C) {
  int i = blockIdx.x * 256 + threadIdx.x;
  if (i >= total) return;
  int c = i % C;
  float t = (x[i] - stats[c]) * rsqrtf(stats[C + c] + 1e-5f) * g[c] + b[c];
  x[i] = (t > 0.f) ? t : (expf(t) - 1.f);
}

__global__ void maxpool_k(const float* __restrict__ x, float* __restrict__ y,
                          int Bn, int L, int C) {
  int Lo = L >> 1;
  int total = Bn * Lo * C;
  int i = blockIdx.x * 256 + threadIdx.x;
  if (i >= total) return;
  int c = i % C;
  int lo = (i / C) % Lo;
  int b = i / (Lo * C);
  float m = -3.4e38f;
  for (int t = -1; t <= 1; ++t) {
    int li = 2 * lo + t;
    if (li >= 0 && li < L) m = fmaxf(m, x[((size_t)b * L + li) * C + c]);
  }
  y[i] = m;
}

__global__ void transpose_k(const float* __restrict__ x, float* __restrict__ y,
                            int Bn, int L, int C) {  // (Bn,L,C) -> (Bn,C,L)
  int total = Bn * L * C;
  int i = blockIdx.x * 256 + threadIdx.x;
  if (i >= total) return;
  int c = i % C;
  int l = (i / C) % L;
  int b = i / (L * C);
  y[((size_t)b * C + c) * L + l] = x[i];
}

__global__ void add_inplace_k(float* __restrict__ y, const float* __restrict__ x,
                              int total) {
  int i = blockIdx.x * 256 + threadIdx.x;
  if (i < total) y[i] += x[i];
}

__global__ void slice_pred_k(const float* __restrict__ pred, float* __restrict__ out,
                             int Bn, int Ld, int outL, int C) {
  int total = Bn * outL * C;
  int i = blockIdx.x * 256 + threadIdx.x;
  if (i >= total) return;
  int c = i % C;
  int t = (i / C) % outL;
  int b = i / (outL * C);
  out[i] = pred[((size_t)b * Ld + (Ld - outL) + t) * C + c];
}

__global__ void cls_head_k(const float* __restrict__ y, const float* __restrict__ w,
                           const float* __restrict__ b, float* __restrict__ out,
                           int Din, int Dout) {
  __shared__ float red[256];
  int bo = blockIdx.x, tid = threadIdx.x;
  int batch = bo / Dout, o = bo % Dout;
  const float* yr = y + (size_t)batch * Din;
  float s = 0.f;
  for (int i = tid; i < Din; i += 256) s += yr[i] * w[(size_t)i * Dout + o];
  red[tid] = s; __syncthreads();
  for (int off = 128; off; off >>= 1) {
    if (tid < off) red[tid] += red[tid + off];
    __syncthreads();
  }
  if (tid == 0) out[batch * Dout + o] = red[0] + b[o];
}

// ---------------------------------------------------------------------------
// Host orchestration
// ---------------------------------------------------------------------------
extern "C" void kernel_launch(void* const* d_in, const int* in_sizes, int n_in,
                              void* d_out, int out_size, void* d_ws, size_t ws_size,
                              hipStream_t stream) {
  (void)in_sizes; (void)n_in; (void)out_size; (void)ws_size;
  const float* x_enc = (const float*)d_in[0];
  const float* x_dec = (const float*)d_in[1];
  auto P = [&](int i) { return (const float*)d_in[i]; };

  // Param flatten order (jax tree: alphabetical keys, depth-first):
  // 2 cls.b 3 cls.w | 4-7 conv_spatial(bn_b,bn_g,conv_b,conv_w)
  // dec0 @8, dec1 @34: cross{k.b,k.w,o.b,o.w,q.b,q.w,v.b,v.w}+0..7,
  //   ff1{b,w}+8,9 ff2{b,w}+10,11 ln1{b,g}+12,13 ln2{b,g}+14,15 ln3{b,g}+16,17,
  //   self{k.b,k.w,o.b,o.w,q.b,q.w,v.b,v.w}+18..25
  // 60 dec_emb_w | 61,62 dec_norm{b,g} | 63-66 enc_conv0 | 67 enc_emb_w
  // enc_s0 @68 (attn 8, ff1 2, ff2 2, ln1 2, ln2 2) | 84,85 enc_s_norm{b,g}
  // enc_t0 @86, enc_t1 @102 | 118,119 enc_t_norm{b,g} | 120 proj.b 121 proj.w

  float* ws = (float*)d_ws;
  const size_t S16 = 4194304;  // 16 MB in floats
  float* BUF0 = ws + 0 * S16;
  float* BUF1 = ws + 1 * S16;
  float* BUF2 = ws + 2 * S16;
  float* BUF3 = ws + 3 * S16;
  float* BUF4 = ws + 4 * S16;
  float* BUF5 = ws + 5 * S16;
  float* BUF6 = ws + 6 * S16;
  float* BUF7 = ws + 7 * S16;
  float* HB   = ws + 8 * S16;          // 8388608 floats
  float* STATS = HB + 8388608;         // 1024 floats

  const int Bn = 64, Ls = 128, Le = 64, Ld = 72, Hh = 8;
  const int M128 = Bn * Ls, M64 = Bn * Le, M72 = Bn * Ld, Msp = Bn * 512;

  auto gemm = [&](const float* A, const float* Bm, float* C, const float* bias,
                  const float* resid, int M, int N, int K, int bks, int bns,
                  int boff, int convL, int convShift, int flags) {
    dim3 g(N / 64, M / 64);
    gemm_wmma_f16<<<g, dim3(256), 0, stream>>>(A, Bm, C, bias, resid, M, N, K,
                                               bks, bns, boff, convL, convShift, flags);
  };
  auto lin = [&](const float* A, int wi, int bi, float* C, int M, int N, int K,
                 int flags, const float* resid) {
    gemm(A, P(wi), C, P(bi), resid, M, N, K, N, 1, 0, 0, 0, flags);
  };
  auto conv3 = [&](const float* A, const float* W, const float* bias, float* C,
                   int M, int Cch, int O, int convL, bool hasBias) {
    int bns = Cch * 3;
    gemm(A, W, C, bias, nullptr, M, O, Cch, 3, bns, 0, convL, -1, 0);
    gemm(A, W, C, bias, nullptr, M, O, Cch, 3, bns, 1, convL, 0, 8);
    gemm(A, W, C, bias, nullptr, M, O, Cch, 3, bns, 2, convL, 1, 8 | (hasBias ? 1 : 0));
  };
  auto lnorm = [&](const float* x, float* y, int gi, int bi, int M, int D) {
    layer_norm_k<<<dim3(M), dim3(128), 0, stream>>>(x, y, P(gi), P(bi), D);
  };
  auto attn_t = [&](const float* Q, const float* K, const float* V, float* O,
                    int Lq, int Lk, int mode, int u, float scale, int bhld) {
    int LqP = (Lq + 31) & ~31, LkP = (Lk + 31) & ~31;
    size_t shm = (size_t)(LqP * 64 + LkP * 64 * 2 + LqP * LkP + LqP * 64 + LqP + 64 + LqP) * 4;
    attn_temporal_k<<<dim3(Bn * Hh), dim3(256), shm, stream>>>(
        Q, K, V, O, Hh, Lq, Lk, LqP, LkP, mode, u, scale, bhld);
  };
  auto elt = [&](int total) { return dim3((total + 255) / 256); };

  // ================= Encoder: temporal branch =================
  conv3(x_enc, P(67), nullptr, BUF0, M128, 512, 512, Ls, false);  // enc embedding
  add_pe_k<<<elt(M128 * 512), 256, 0, stream>>>(BUF0, M128 * 512, Ls, 512);

  // enc_t0
  lin(BUF0, 91, 90, BUF1, M128, 512, 512, 1, nullptr);            // q
  lin(BUF0, 87, 86, BUF2, M128, 512, 512, 1, nullptr);            // k
  lin(BUF0, 93, 92, BUF3, M128, 512, 512, 1, nullptr);            // v
  attn_t(BUF1, BUF2, BUF3, BUF4, Ls, Ls, 1, 25, 0.125f, 0);
  lin(BUF4, 89, 88, BUF5, M128, 512, 512, 1 | 4, BUF0);           // o + resid
  lnorm(BUF5, BUF5, 99, 98, M128, 512);                           // ln1
  for (int c0 = 0; c0 < M128; c0 += 4096) {
    lin(BUF5 + (size_t)c0 * 512, 95, 94, HB, 4096, 2048, 512, 1 | 2, nullptr);
    lin(HB, 97, 96, BUF0 + (size_t)c0 * 512, 4096, 512, 2048, 1 | 4,
        BUF5 + (size_t)c0 * 512);
  }
  lnorm(BUF0, BUF0, 101, 100, M128, 512);                         // ln2

  // enc_conv0 distilling layer
  conv3(BUF0, P(66), P(65), BUF1, M128, 512, 512, Ls, true);
  bn_stats_k<<<dim3(512), dim3(256), 0, stream>>>(BUF1, STATS, M128, 512);
  bn_elu_k<<<elt(M128 * 512), 256, 0, stream>>>(BUF1, STATS, P(64), P(63), M128 * 512, 512);
  maxpool_k<<<elt(M64 * 512), 256, 0, stream>>>(BUF1, BUF2, Bn, Ls, 512);

  // enc_t1 (L = 64)
  lin(BUF2, 107, 106, BUF1, M64, 512, 512, 1, nullptr);
  lin(BUF2, 103, 102, BUF3, M64, 512, 512, 1, nullptr);
  lin(BUF2, 109, 108, BUF5, M64, 512, 512, 1, nullptr);
  attn_t(BUF1, BUF3, BUF5, BUF6, Le, Le, 1, 25, 0.125f, 0);
  lin(BUF6, 105, 104, BUF7, M64, 512, 512, 1 | 4, BUF2);
  lnorm(BUF7, BUF7, 115, 114, M64, 512);
  lin(BUF7, 111, 110, HB, M64, 2048, 512, 1 | 2, nullptr);
  lin(HB, 113, 112, BUF0, M64, 512, 2048, 1 | 4, BUF7);
  lnorm(BUF0, BUF0, 117, 116, M64, 512);
  lnorm(BUF0, BUF0, 119, 118, M64, 512);                          // enc_t_norm

  // ================= Encoder: spatial branch =================
  transpose_k<<<elt(M128 * 512), 256, 0, stream>>>(x_enc, BUF1, Bn, Ls, 512);  // (B,512,128)
  lin(BUF1, 73, 72, BUF2, Msp, 128, 128, 1, nullptr);             // q
  lin(BUF1, 69, 68, BUF3, Msp, 128, 128, 1, nullptr);             // k
  lin(BUF1, 75, 74, BUF5, Msp, 128, 128, 1, nullptr);             // v
  {
    size_t shm = (size_t)(512 * 32 * 2 + 512 * 16 + 32 * 512 + 1024 + 512 + 16 + 1024) * 4;
    attn_spatial_k<<<dim3(Bn * Hh), dim3(256), shm, stream>>>(BUF2, BUF3, BUF5, BUF6,
                                                              Hh, 512, 70, 0.25f);
  }
  lin(BUF6, 71, 70, BUF7, Msp, 128, 128, 1 | 4, BUF1);            // o + resid
  lnorm(BUF7, BUF7, 81, 80, Msp, 128);
  for (int c0 = 0; c0 < Msp; c0 += 8192) {
    lin(BUF7 + (size_t)c0 * 128, 77, 76, HB, 8192, 1024, 128, 1 | 2, nullptr);
    lin(HB, 79, 78, BUF1 + (size_t)c0 * 128, 8192, 128, 1024, 1 | 4,
        BUF7 + (size_t)c0 * 128);
  }
  lnorm(BUF1, BUF1, 83, 82, Msp, 128);
  lnorm(BUF1, BUF1, 85, 84, Msp, 128);                            // enc_s_norm
  transpose_k<<<elt(M128 * 512), 256, 0, stream>>>(BUF1, BUF2, Bn, 512, 128);  // back to (B,128,512)
  conv3(BUF2, P(7), P(6), BUF3, M128, 512, 512, Ls, true);        // conv_spatial
  bn_stats_k<<<dim3(512), dim3(256), 0, stream>>>(BUF3, STATS, M128, 512);
  bn_elu_k<<<elt(M128 * 512), 256, 0, stream>>>(BUF3, STATS, P(5), P(4), M128 * 512, 512);
  maxpool_k<<<elt(M64 * 512), 256, 0, stream>>>(BUF3, BUF4, Bn, Ls, 512);
  add_inplace_k<<<elt(M64 * 512), 256, 0, stream>>>(BUF4, BUF0, M64 * 512);  // enc_out = BUF4

  // ================= Decoder =================
  conv3(x_dec, P(60), nullptr, BUF0, M72, 512, 512, Ld, false);
  add_pe_k<<<elt(M72 * 512), 256, 0, stream>>>(BUF0, M72 * 512, Ld, 512);

  float* cur = BUF0;
  float* oth = BUF6;
  for (int li = 0; li < 2; ++li) {
    int b0 = (li == 0) ? 8 : 34;
    // self (ProbSparse causal, mix=True -> BHLD)
    lin(cur, b0 + 23, b0 + 22, BUF1, M72, 512, 512, 1, nullptr);
    lin(cur, b0 + 19, b0 + 18, BUF2, M72, 512, 512, 1, nullptr);
    lin(cur, b0 + 25, b0 + 24, BUF3, M72, 512, 512, 1, nullptr);
    attn_t(BUF1, BUF2, BUF3, BUF5, Ld, Ld, 2, 25, 0.125f, 1);
    lin(BUF5, b0 + 21, b0 + 20, oth, M72, 512, 512, 1 | 4, cur);
    lnorm(oth, oth, b0 + 13, b0 + 12, M72, 512);                  // ln1
    // cross (full attention against enc_out)
    lin(oth, b0 + 5, b0 + 4, BUF1, M72, 512, 512, 1, nullptr);
    lin(BUF4, b0 + 1, b0 + 0, BUF2, M64, 512, 512, 1, nullptr);
    lin(BUF4, b0 + 7, b0 + 6, BUF3, M64, 512, 512, 1, nullptr);
    attn_t(BUF1, BUF2, BUF3, BUF5, Ld, Le, 0, 0, 0.125f, 0);
    lin(BUF5, b0 + 3, b0 + 2, cur, M72, 512, 512, 1 | 4, oth);
    lnorm(cur, cur, b0 + 15, b0 + 14, M72, 512);                  // ln2
    for (int c0 = 0; c0 < M72; c0 += 2304) {
      lin(cur + (size_t)c0 * 512, b0 + 9, b0 + 8, HB, 2304, 2048, 512, 1 | 2, nullptr);
      lin(HB, b0 + 11, b0 + 10, oth + (size_t)c0 * 512, 2304, 512, 2048, 1 | 4,
          cur + (size_t)c0 * 512);
    }
    lnorm(oth, oth, b0 + 17, b0 + 16, M72, 512);                  // ln3
    float* t = cur; cur = oth; oth = t;
  }
  lnorm(cur, cur, 62, 61, M72, 512);                              // dec_norm

  // ================= Heads =================
  lin(cur, 121, 120, BUF1, M72, 512, 512, 1, nullptr);            // proj
  float* out = (float*)d_out;
  slice_pred_k<<<elt(Bn * 24 * 512), 256, 0, stream>>>(BUF1, out, Bn, Ld, 24, 512);
  cls_head_k<<<dim3(Bn * 4), dim3(256), 0, stream>>>(cur, P(3), P(2),
                                                     out + (size_t)Bn * 24 * 512,
                                                     Ld * 512, 4);
}
